// FullAttention_15994458210525
// MI455X (gfx1250) — matrix operand order repaired and checked by reading, not compiled, not run on hardware
//
#include <hip/hip_runtime.h>

typedef __attribute__((ext_vector_type(16))) _Float16 v16h;
typedef __attribute__((ext_vector_type(8)))  float    v8f;

#define B_  4
#define L_  2048
#define H_  8
#define E_  64
#define HE_ (H_ * E_)   // 512 floats between consecutive sequence positions

// DPP butterfly max over each independent 16-lane half (rows live in halves).
// quad_perm(xor1)=0xB1, quad_perm(xor2)=0x4E, row_half_mirror=0x141, row_mirror=0x140.
#define DPP_MAX_STEP(x, ctrl)                                                   \
    do {                                                                        \
        int _xi = __builtin_bit_cast(int, (x));                                 \
        int _yi = __builtin_amdgcn_update_dpp(_xi, _xi, (ctrl), 0xf, 0xf, false);\
        (x) = fmaxf((x), __builtin_bit_cast(float, _yi));                       \
    } while (0)

// Flash-attention forward, causal, fp32 in/out, f16 WMMA compute with f32 accum.
// One workgroup = 4 waves = 64 query rows of one (b,h); each wave owns 16 rows.
// K/V staged cooperatively into LDS as f16 (V transposed), 32 keys per step.
// Softmax runs in the exp2 domain (v_exp_f32 is exp2); row sums of P come from a
// 5th WMMA against an all-ones B operand; the alpha-rescale is skipped via wave
// vote whenever the running max is unchanged. All B operands of a WMMA group are
// preloaded into distinct registers so DS waits overlap the matrix pipe.
__global__ __launch_bounds__(128) void fattn_fwd_kernel(const float* __restrict__ qg,
                                                        const float* __restrict__ kg,
                                                        const float* __restrict__ vg,
                                                        float* __restrict__ og) {
    __shared__ __align__(16) _Float16 Kbuf[32 * 64];     // [key][e]       row-major
    __shared__ __align__(16) _Float16 Vbuf[64 * 32];     // [d][key]       transposed
    __shared__ __align__(16) _Float16 Pbuf[4][16 * 32];  // per-wave [m][k]

    const int tid  = threadIdx.x;
    const int wave = tid >> 5;
    const int lane = tid & 31;
    const int blk  = blockIdx.x;        // 64-row query block
    const int bh   = blockIdx.y;
    const int b    = bh >> 3;           // / H_
    const int h    = bh & 7;            // % H_

    const size_t base = (size_t)b * L_ * HE_ + (size_t)h * E_;
    const float* Qb = qg + base;
    const float* Kb = kg + base;
    const float* Vb = vg + base;
    float*       Ob = og + base;

    const int q0blk = blk * 64;
    const int q0    = q0blk + wave * 16;   // this wave's first query row

    const int m  = lane & 15;              // row/col index within 16-lane half
    const int hi = lane >> 4;              // which half of the wave

    // ---- Load Q tile once as two 16x32 f16 A-operands (e 0..31, 32..63) ----
    // A operand in this code's contiguous k order: lane l(<16): M=l, K 0..15; lane l+16: same M, K 16..31.
    v16h qa0, qa1;
    {
        const float* rowp = Qb + (size_t)(q0 + m) * HE_;
        #pragma unroll
        for (int i = 0; i < 16; ++i) {
            const int kk = i + 16 * hi;
            qa0[i] = (_Float16)rowp[kk];
            qa1[i] = (_Float16)rowp[kk + 32];
        }
    }

    // All-ones B operand: P(16x32) x ones(32x16) -> row sums replicated per column.
    v16h ones_b;
    #pragma unroll
    for (int i = 0; i < 16; ++i) ones_b[i] = (_Float16)1.0f;

    // exp2-domain scale: 1/sqrt(E) * log2(e); softmax is base-invariant.
    const float scale2 = 0.125f * 1.44269504088896340736f;

    float m_i[8];
    v8f   acc[4], l_acc;
    #pragma unroll
    for (int r = 0; r < 8; ++r) { m_i[r] = -1e30f; l_acc[r] = 0.0f; }
    #pragma unroll
    for (int d = 0; d < 4; ++d) {
        #pragma unroll
        for (int i = 0; i < 8; ++i) acc[d][i] = 0.0f;
    }

    const int jend = q0blk + 64;           // causal: keys < end of this 64-row block
    for (int j = 0; j < jend; j += 32) {
        // ---- cooperative staging: 32x64 fp32 -> f16 LDS (K row-major, V transposed)
        #pragma unroll
        for (int it = 0; it < 4; ++it) {
            const int idx = tid + it * 128;      // 512 float4 slots
            const int row = idx >> 4;            // key 0..31
            const int c4  = idx & 15;            // float4 column
            const float4 kq = *(const float4*)(Kb + (size_t)(j + row) * HE_ + c4 * 4);
            _Float16* kd = &Kbuf[row * 64 + c4 * 4];
            kd[0] = (_Float16)kq.x; kd[1] = (_Float16)kq.y;
            kd[2] = (_Float16)kq.z; kd[3] = (_Float16)kq.w;
            const float4 vq = *(const float4*)(Vb + (size_t)(j + row) * HE_ + c4 * 4);
            Vbuf[(c4 * 4 + 0) * 32 + row] = (_Float16)vq.x;
            Vbuf[(c4 * 4 + 1) * 32 + row] = (_Float16)vq.y;
            Vbuf[(c4 * 4 + 2) * 32 + row] = (_Float16)vq.z;
            Vbuf[(c4 * 4 + 3) * 32 + row] = (_Float16)vq.w;
            if (j + 32 < jend) {   // prefetch next tile -> global_prefetch_b8
                __builtin_prefetch(Kb + (size_t)(j + 32 + row) * HE_ + c4 * 4, 0, 1);
                __builtin_prefetch(Vb + (size_t)(j + 32 + row) * HE_ + c4 * 4, 0, 1);
            }
        }
        __syncthreads();

        if (j <= q0 + 15) {   // wave has at least one unmasked key in this tile
            // ---- preload all 4 K B-operands (distinct regs -> overlapped waits)
            v16h kb[2][2];
            #pragma unroll
            for (int c = 0; c < 2; ++c) {
                const int kr = c * 16 + m;     // B, contiguous k as Q: lane half selects K 0..15 / 16..31
                #pragma unroll
                for (int i = 0; i < 16; ++i) {
                    kb[c][0][i] = Kbuf[kr * 64 + hi * 16 + i];
                    kb[c][1][i] = Kbuf[kr * 64 + 32 + hi * 16 + i];
                }
            }
            // ---- S(16x32) = Q K^T : two 16-col chunks, K-dim 64 = 2 WMMAs each
            v8f sc[2];
            #pragma unroll
            for (int c = 0; c < 2; ++c) {
                v8f s;
                #pragma unroll
                for (int i = 0; i < 8; ++i) s[i] = 0.0f;
                s = __builtin_amdgcn_wmma_f32_16x16x32_f16(false, qa0, false, kb[c][0],
                                                           (short)0, s, false, false);
                s = __builtin_amdgcn_wmma_f32_16x16x32_f16(false, qa1, false, kb[c][1],
                                                           (short)0, s, false, false);
                sc[c] = s;
            }
            // ---- scale; causal mask only on diagonal tiles (wave-uniform branch)
            if (j + 31 > q0) {
                #pragma unroll
                for (int c = 0; c < 2; ++c) {
                    #pragma unroll
                    for (int r = 0; r < 8; ++r) {
                        const int qrow = q0 + r + 8 * hi;
                        const int key  = j + c * 16 + m;
                        const float sv = sc[c][r] * scale2;
                        sc[c][r] = (key > qrow) ? -1e30f : sv;
                    }
                }
            } else {
                #pragma unroll
                for (int c = 0; c < 2; ++c) {
                    #pragma unroll
                    for (int r = 0; r < 8; ++r) sc[c][r] *= scale2;
                }
            }
            // ---- row max via DPP butterfly (pure VALU, no LDS traffic)
            float rmax[8];
            #pragma unroll
            for (int r = 0; r < 8; ++r) rmax[r] = fmaxf(sc[0][r], sc[1][r]);
            #pragma unroll
            for (int r = 0; r < 8; ++r) {
                DPP_MAX_STEP(rmax[r], 0xB1);   // quad_perm xor1
                DPP_MAX_STEP(rmax[r], 0x4E);   // quad_perm xor2
                DPP_MAX_STEP(rmax[r], 0x141);  // row_half_mirror
                DPP_MAX_STEP(rmax[r], 0x140);  // row_mirror
            }
            // ---- rescale only when the running max moved (wave-uniform vote)
            int changed = 0;
            #pragma unroll
            for (int r = 0; r < 8; ++r) changed |= (rmax[r] > m_i[r]) ? 1 : 0;
            if (__any(changed)) {
                #pragma unroll
                for (int r = 0; r < 8; ++r) {
                    const float mn = fmaxf(m_i[r], rmax[r]);
                    const float a  = __builtin_amdgcn_exp2f(m_i[r] - mn);
                    m_i[r] = mn;
                    l_acc[r] *= a;
                    #pragma unroll
                    for (int d = 0; d < 4; ++d) acc[d][r] *= a;
                }
            }
            // ---- P = exp2(S - m)
            float p0[8], p1[8];
            #pragma unroll
            for (int r = 0; r < 8; ++r) {
                p0[r] = __builtin_amdgcn_exp2f(sc[0][r] - m_i[r]);
                p1[r] = __builtin_amdgcn_exp2f(sc[1][r] - m_i[r]);
            }
            // ---- redistribute P: C-layout -> LDS [m][k] -> A-layout operand
            _Float16* pw = Pbuf[wave];
            #pragma unroll
            for (int r = 0; r < 8; ++r) {
                pw[(r + 8 * hi) * 32 + m]      = (_Float16)p0[r];
                pw[(r + 8 * hi) * 32 + 16 + m] = (_Float16)p1[r];
            }
            asm volatile("s_wait_dscnt 0" ::: "memory");  // wave-local LDS RAW order
            v16h pa;
            #pragma unroll
            for (int i = 0; i < 16; ++i) {
                const int kk = i + 16 * hi;
                pa[i] = pw[m * 32 + kk];
            }
            // ---- preload all 4 V B-operands (distinct regs -> overlapped waits)
            v16h vb[4];
            #pragma unroll
            for (int d = 0; d < 4; ++d) {
                #pragma unroll
                for (int i = 0; i < 16; ++i)
                    vb[d][i] = Vbuf[(d * 16 + m) * 32 + hi * 16 + i];
            }
            // ---- l += P*ones (row sums, replicated across columns)
            l_acc = __builtin_amdgcn_wmma_f32_16x16x32_f16(false, pa, false, ones_b,
                                                           (short)0, l_acc, false, false);
            // ---- O(16x64) += P(16x32) * V(32x64): 4 WMMAs
            #pragma unroll
            for (int d = 0; d < 4; ++d) {
                acc[d] = __builtin_amdgcn_wmma_f32_16x16x32_f16(false, pa, false, vb[d],
                                                                (short)0, acc[d], false, false);
            }
        }
        __syncthreads();
    }

    // ---- epilogue: O = acc / l, fp32 out, layout [B,L,H,E]
    #pragma unroll
    for (int r = 0; r < 8; ++r) {
        const float inv = 1.0f / l_acc[r];
        #pragma unroll
        for (int d = 0; d < 4; ++d) {
            Ob[(size_t)(q0 + r + 8 * hi) * HE_ + d * 16 + m] = acc[d][r] * inv;
        }
    }
}

extern "C" void kernel_launch(void* const* d_in, const int* in_sizes, int n_in,
                              void* d_out, int out_size, void* d_ws, size_t ws_size,
                              hipStream_t stream) {
    const float* q = (const float*)d_in[0];
    const float* k = (const float*)d_in[1];
    const float* v = (const float*)d_in[2];
    float*       o = (float*)d_out;
    dim3 grid(L_ / 64, B_ * H_);   // (32, 32) workgroups
    fattn_fwd_kernel<<<grid, 128, 0, stream>>>(q, k, v, o);
}
